// Model_37374805410182
// MI455X (gfx1250) — matrix operand order repaired and checked by reading, not compiled, not run on hardware
//
#include <hip/hip_runtime.h>
#include <hip/hip_bf16.h>

// Problem dims
#define BB    512
#define TT    256
#define DIN   400
#define DINP  416          // 13 * 32, zero-padded K for bf16 WMMA
#define HH    128
#define G4    512          // 4*H
#define BT    (BB*TT)      // 131072

typedef __attribute__((ext_vector_type(16))) __bf16 bf16x16;
typedef __attribute__((ext_vector_type(8)))  __bf16 bf16x8;
typedef __attribute__((ext_vector_type(8)))  float  f32x8;
typedef __attribute__((ext_vector_type(4)))  unsigned int u32x4;
typedef __attribute__((ext_vector_type(8)))  int i32x8;
typedef __attribute__((ext_vector_type(4)))  int i32x4;

// workspace layout (byte offsets, all 256B aligned)
#define WIH_OFF  0u          // 512*416*2 = 425984
#define WHH_OFF  425984u     // 512*128*2 = 131072
#define W1_OFF   557056u     // 128*128*2 = 32768
#define BIAS_OFF 589824u     // 512*4    = 2048
#define XG_OFF   1048576u    // 131072*512*2 = 134217728

#if defined(__gfx1250__) && __has_builtin(__builtin_amdgcn_tensor_load_to_lds) && \
    __has_builtin(__builtin_amdgcn_s_wait_tensorcnt)
#define HAS_TDM 1
#else
#define HAS_TDM 0
#endif

__device__ __forceinline__ f32x8 wmma_bf16(bf16x16 a, bf16x16 b, f32x8 c) {
  return __builtin_amdgcn_wmma_f32_16x16x32_bf16(false, a, false, b, (short)0, c,
                                                 false, false);
}

// ---------------- kernel 0: weight conversion fp32 -> bf16 ----------------
__global__ void k_convert(const float* __restrict__ wih,
                          const float* __restrict__ whh,
                          const float* __restrict__ bih,
                          const float* __restrict__ bhh,
                          const float* __restrict__ w1,
                          char* __restrict__ ws) {
  __bf16* wih_bf = (__bf16*)(ws + WIH_OFF);
  __bf16* whh_bf = (__bf16*)(ws + WHH_OFF);
  __bf16* w1_bf  = (__bf16*)(ws + W1_OFF);
  float*  biasf  = (float*)(ws + BIAS_OFF);
  const int NWIH = G4 * DINP;   // 212992
  const int NWHH = G4 * HH;     // 65536
  const int NW1  = HH * HH;     // 16384
  int i = blockIdx.x * 256 + threadIdx.x;
  if (i < NWIH) {
    int r = i / DINP, c = i % DINP;
    wih_bf[i] = (c < DIN) ? (__bf16)wih[r * DIN + c] : (__bf16)0.f;
  } else if (i < NWIH + NWHH) {
    int j = i - NWIH;
    whh_bf[j] = (__bf16)whh[j];
  } else if (i < NWIH + NWHH + NW1) {
    int j = i - NWIH - NWHH;
    w1_bf[j] = (__bf16)w1[j];
  } else if (i < NWIH + NWHH + NW1 + G4) {
    int j = i - NWIH - NWHH - NW1;
    biasf[j] = bih[j] + bhh[j];
  }
}

// ------------- kernel 1: xg = x @ Wih^T + (bih+bhh), bf16 out -------------
// one wave -> 16(M) x 256(N) strip; A built on the fly from fp32 x.
// B fragments addressed as single base + compile-time immediate offsets.
__global__ void __launch_bounds__(256)
k_xgemm(const float* __restrict__ x, const char* __restrict__ ws,
        __bf16* __restrict__ xg) {
  const __bf16* wih_bf = (const __bf16*)(ws + WIH_OFF);
  const float*  biasf  = (const float*)(ws + BIAS_OFF);
  const int lane = threadIdx.x & 31;
  const int wid  = blockIdx.x * 8 + (threadIdx.x >> 5);  // 0..16383
  const int nstrip = wid & 1;
  const int mtile  = wid >> 1;                           // 0..8191
  const int n0 = nstrip * 256;
  const int lm = lane & 15, lh = lane >> 4;
  const size_t rowoff = (size_t)(mtile * 16 + lm) * DIN;
  const float*  abase = x + rowoff + lh * 16;
  const __bf16* bbase = wih_bf + (n0 + lm) * DINP + lh * 16;
  f32x8 acc[16] = {};
  // 12 regular K-chunks: all loads base + immediate offset, no clamps
  #pragma unroll 1
  for (int kc = 0; kc < 12; ++kc) {
    const int kb = kc * 32;
    __builtin_prefetch(abase + kb + 128, 0, 1);   // global_prefetch_b8
    f32x8 lo = *(const f32x8*)(abase + kb);
    f32x8 hi = *(const f32x8*)(abase + kb + 8);
    bf16x16 a;
    #pragma unroll
    for (int u = 0; u < 8; ++u) { a[u] = (__bf16)lo[u]; a[8 + u] = (__bf16)hi[u]; }
    #pragma unroll
    for (int nt = 0; nt < 16; ++nt)
      acc[nt] = wmma_bf16(a, *(const bf16x16*)(bbase + nt * (16 * DINP) + kb),
                          acc[nt]);
  }
  // K tail (kc=12, cols 384..415; 400..415 hit zero weights): clamp reads
  {
    const size_t xlim = (size_t)BT * DIN - 8;
    size_t o0 = rowoff + 384 + lh * 16;
    size_t o1 = o0 + 8;
    if (o0 > xlim) o0 = xlim;
    if (o1 > xlim) o1 = xlim;
    f32x8 lo = *(const f32x8*)(x + o0);
    f32x8 hi = *(const f32x8*)(x + o1);
    bf16x16 a;
    #pragma unroll
    for (int u = 0; u < 8; ++u) { a[u] = (__bf16)lo[u]; a[8 + u] = (__bf16)hi[u]; }
    #pragma unroll
    for (int nt = 0; nt < 16; ++nt)
      acc[nt] = wmma_bf16(a, *(const bf16x16*)(bbase + nt * (16 * DINP) + 384),
                          acc[nt]);
  }
  #pragma unroll
  for (int nt = 0; nt < 16; ++nt) {
    const int col = n0 + nt * 16 + lm;
    const float bv = biasf[col];
    #pragma unroll
    for (int r = 0; r < 8; ++r) {
      const int m = mtile * 16 + r + 8 * lh;
      xg[(size_t)m * G4 + col] = (__bf16)(acc[nt][r] + bv);
    }
  }
}

// ---- kernel 2: LSTM scan over T (16 batch rows / block) + MLP head ----
// TDM double-buffers the next step's 16x512 bf16 xg tile into LDS while the
// current step's h@Whh^T WMMAs run.
__global__ void __launch_bounds__(256)
k_scan(const __bf16* __restrict__ xg, const char* __restrict__ ws,
       const float* __restrict__ b1, const float* __restrict__ w2,
       const float* __restrict__ b2, float* __restrict__ out) {
  const __bf16* whh_bf = (const __bf16*)(ws + WHH_OFF);
  const __bf16* w1_bf  = (const __bf16*)(ws + W1_OFF);
  __shared__ __bf16 gates_s[16 * G4];   // 16 KB (bf16 exchange buffer)
  __shared__ float  c_s[16 * HH];       // 8 KB (fp32 cell state)
  __shared__ __bf16 h_s[16 * HH];       // 4 KB
#if HAS_TDM
  __shared__ __bf16 xgbuf[2][16 * G4];  // 32 KB double buffer for TDM
#endif
  const int tid  = threadIdx.x;
  const int lane = tid & 31, wave = tid >> 5;
  const int lm = lane & 15, lh = lane >> 4;
  const int brow0 = blockIdx.x * 16;

  // Whh B-fragments resident in VGPRs for the entire scan (reused 256x)
  bf16x16 wf[4][4];
  #pragma unroll
  for (int kc = 0; kc < 4; ++kc)
    #pragma unroll
    for (int nt = 0; nt < 4; ++nt) {
      const int n = wave * 64 + nt * 16 + lm;
      wf[kc][nt] = *(const bf16x16*)(whh_bf + n * HH + kc * 32 + lh * 16);
    }
  // h0 = c0 = 0
  #pragma unroll
  for (int u = 0; u < 8; ++u) {
    h_s[tid * 8 + u] = (__bf16)0.f;
    c_s[tid * 8 + u] = 0.f;
  }

#if HAS_TDM
  // Constant D# group1: data_size=2B; tensor/tile dim0=512; dim1=16 rows;
  // dim0_stride = T*4H = 131072 elements.
  const i32x8 g1 = {0x00010000, 0x02000000, 0x00100000, 0x02000000,
                    16, 131072, 0, 0};
  const i32x4 z4 = {0, 0, 0, 0};
#if defined(__clang_major__) && __clang_major__ >= 23
  const i32x8 z8 = {0, 0, 0, 0, 0, 0, 0, 0};
#define TDM_LOAD(g0) __builtin_amdgcn_tensor_load_to_lds((g0), g1, z4, z4, z8, 0)
#else
#define TDM_LOAD(g0) __builtin_amdgcn_tensor_load_to_lds((g0), g1, z4, z4, 0)
#endif
  if (wave == 0) {  // prologue: stage xg tile for t=0 into buffer 0
    const uint64_t ga = (uint64_t)(uintptr_t)(xg + (size_t)brow0 * TT * G4);
    u32x4 g0;
    g0[0] = 1u;                                             // count=1
    g0[1] = (unsigned)(uintptr_t)(&xgbuf[0][0]);            // LDS byte addr
    g0[2] = (unsigned)ga;
    g0[3] = (unsigned)((ga >> 32) & 0x01FFFFFFu) | (2u << 30);  // type=2
    TDM_LOAD(g0);
  }
#endif
  __syncthreads();

  const int mm = tid >> 4;            // row 0..15
  const int j0 = (tid & 15) * 8;      // col start in H
  const size_t xrow = (size_t)(brow0 + mm) * TT;
  (void)xrow;

  #pragma unroll 1
  for (int t = 0; t < TT; ++t) {
#if HAS_TDM
    if (wave == 0 && t + 1 < TT) {    // kick DMA for step t+1
      const uint64_t ga =
          (uint64_t)(uintptr_t)(xg + ((size_t)brow0 * TT + (t + 1)) * G4);
      u32x4 g0;
      g0[0] = 1u;
      g0[1] = (unsigned)(uintptr_t)(&xgbuf[(t + 1) & 1][0]);
      g0[2] = (unsigned)ga;
      g0[3] = (unsigned)((ga >> 32) & 0x01FFFFFFu) | (2u << 30);
      TDM_LOAD(g0);
    }
#else
    const size_t xb = (xrow + t) * G4;  // hoisted: overlap with WMMA phase
    bf16x8 rxi  = *(const bf16x8*)(xg + xb + j0);
    bf16x8 rxf  = *(const bf16x8*)(xg + xb + 128 + j0);
    bf16x8 rxg  = *(const bf16x8*)(xg + xb + 256 + j0);
    bf16x8 rxo  = *(const bf16x8*)(xg + xb + 384 + j0);
#endif
    // gates_tile = h @ Whh^T   (wave covers cols [wave*64, wave*64+63])
    f32x8 acc[4] = {};
    #pragma unroll
    for (int kc = 0; kc < 4; ++kc) {
      bf16x8 alo = *(const bf16x8*)(h_s + lm * HH + kc * 32 + lh * 16);
      bf16x8 ahi = *(const bf16x8*)(h_s + lm * HH + kc * 32 + 8 + lh * 16);
      bf16x16 a;
      #pragma unroll
      for (int u = 0; u < 8; ++u) { a[u] = alo[u]; a[8 + u] = ahi[u]; }
      #pragma unroll
      for (int nt = 0; nt < 4; ++nt)
        acc[nt] = wmma_bf16(a, wf[kc][nt], acc[nt]);
    }
    #pragma unroll
    for (int nt = 0; nt < 4; ++nt) {
      const int col = wave * 64 + nt * 16 + lm;
      #pragma unroll
      for (int r = 0; r < 8; ++r)
        gates_s[(r + 8 * lh) * G4 + col] = (__bf16)acc[nt][r];
    }
#if HAS_TDM
    if (wave == 0) {  // retire DMA for step t; keep t+1 in flight
      if (t + 1 < TT) __builtin_amdgcn_s_wait_tensorcnt(1);
      else            __builtin_amdgcn_s_wait_tensorcnt(0);
    }
#endif
    __syncthreads();
    // gate nonlinearities + state update (PyTorch i,f,g,o order)
#if HAS_TDM
    const __bf16* xp = &xgbuf[t & 1][mm * G4];
    bf16x8 rxi = *(const bf16x8*)(xp + j0);
    bf16x8 rxf = *(const bf16x8*)(xp + 128 + j0);
    bf16x8 rxg = *(const bf16x8*)(xp + 256 + j0);
    bf16x8 rxo = *(const bf16x8*)(xp + 384 + j0);
#endif
    #pragma unroll
    for (int u = 0; u < 8; ++u) {
      const int j = j0 + u;
      float gi = (float)gates_s[mm * G4 + j]       + (float)rxi[u];
      float gf = (float)gates_s[mm * G4 + 128 + j] + (float)rxf[u];
      float gg = (float)gates_s[mm * G4 + 256 + j] + (float)rxg[u];
      float go = (float)gates_s[mm * G4 + 384 + j] + (float)rxo[u];
      float si = 1.f / (1.f + __expf(-gi));
      float sf = 1.f / (1.f + __expf(-gf));
      float tg = tanhf(gg);
      float so = 1.f / (1.f + __expf(-go));
      float cv = sf * c_s[mm * HH + j] + si * tg;
      c_s[mm * HH + j] = cv;
      h_s[mm * HH + j] = (__bf16)(so * tanhf(cv));
    }
    __syncthreads();
  }

  // ---- head: z = relu(h @ W1^T + b1); out = z @ W2^T + b2 ----
  f32x8 zacc = {};
  #pragma unroll
  for (int kc = 0; kc < 4; ++kc) {
    bf16x8 alo = *(const bf16x8*)(h_s + lm * HH + kc * 32 + lh * 16);
    bf16x8 ahi = *(const bf16x8*)(h_s + lm * HH + kc * 32 + 8 + lh * 16);
    bf16x16 a;
    #pragma unroll
    for (int u = 0; u < 8; ++u) { a[u] = alo[u]; a[8 + u] = ahi[u]; }
    const int n = wave * 16 + lm;   // each wave owns one 16-col N tile of z
    bf16x16 b = *(const bf16x16*)(w1_bf + n * HH + kc * 32 + lh * 16);
    zacc = wmma_bf16(a, b, zacc);
  }
  __bf16* z_s = gates_s;            // reuse exchange LDS
  {
    const int col = wave * 16 + lm;
    const float bv = b1[col];
    #pragma unroll
    for (int r = 0; r < 8; ++r) {
      float zv = zacc[r] + bv;
      z_s[(r + 8 * lh) * HH + col] = (__bf16)(zv > 0.f ? zv : 0.f);
    }
  }
  __syncthreads();
  if (tid < 48) {
    const int m = tid / 3, q = tid % 3;
    float s = b2[q];
    #pragma unroll 4
    for (int j = 0; j < HH; ++j)
      s += (float)z_s[m * HH + j] * w2[q * HH + j];
    out[(brow0 + m) * 3 + q] = s;
  }
}

extern "C" void kernel_launch(void* const* d_in, const int* in_sizes, int n_in,
                              void* d_out, int out_size, void* d_ws, size_t ws_size,
                              hipStream_t stream) {
  const float* x   = (const float*)d_in[0];
  const float* wih = (const float*)d_in[1];
  const float* whh = (const float*)d_in[2];
  const float* bih = (const float*)d_in[3];
  const float* bhh = (const float*)d_in[4];
  const float* w1  = (const float*)d_in[5];
  const float* b1  = (const float*)d_in[6];
  const float* w2  = (const float*)d_in[7];
  const float* b2  = (const float*)d_in[8];
  char* ws = (char*)d_ws;
  __bf16* xg = (__bf16*)(ws + XG_OFF);
  float* out = (float*)d_out;

  // 0) weights -> bf16 (+K pad), bias fold
  k_convert<<<1154, 256, 0, stream>>>(wih, whh, bih, bhh, w1, ws);
  // 1) xg = x @ Wih^T + bias : 8192 M-tiles * 2 N-strips = 16384 waves
  k_xgemm<<<2048, 256, 0, stream>>>(x, ws, xg);
  // 2) sequential LSTM scan (TDM-prefetched xg) + MLP head : 32 batch tiles
  k_scan<<<32, 256, 0, stream>>>(xg, ws, b1, w2, b2, out);
}
